// CrossAttention_18468359373399
// MI455X (gfx1250) — matrix operand order repaired and checked by reading, not compiled, not run on hardware
//
#include <hip/hip_runtime.h>
#include <hip/hip_bf16.h>

typedef __attribute__((ext_vector_type(16))) _Float16 v16h;
typedef __attribute__((ext_vector_type(8)))  _Float16 v8h;
typedef __attribute__((ext_vector_type(8)))  float    v8f;
typedef __attribute__((ext_vector_type(4)))  unsigned int u32x4;
typedef __attribute__((ext_vector_type(8)))  int i32x8;
typedef __attribute__((ext_vector_type(4)))  int i32x4;

// address-space-qualified int4 for the async-LDS builtin (sig from compiler diag:
// param0 = int4 addrspace(1)*, param1 = int4 addrspace(3)*, imm offset, imm cpol)
typedef __attribute__((address_space(1))) i32x4 gas_i32x4;
typedef __attribute__((address_space(3))) i32x4 las_i32x4;

#define BB   8
#define SQ   2048
#define SKV  1024
#define DE   512
#define DC   768
#define NH   8
#define DH   64
#define SPITCH 1032   // 1024 + 8 pad

// LDS layout for the attention kernel (bytes)
#define OFF_S     0
#define OFF_PS    (16 * SPITCH * 4)              //  66048
#define OFF_QS    (OFF_PS   + 16 * SPITCH * 2)   //  99072
#define OFF_OACC  (OFF_QS   + 16 * DH * 2)       // 101120
#define OFF_VS    (OFF_OACC + 16 * DH * 4)       // 105216  (8 waves x 32x64 f16)
#define SMEM_TOTAL (OFF_VS + 8 * 32 * DH * 2)    // 137984

// ---- CDNA5 feature probes (device pass only) ----
#if defined(__AMDGCN__) && __has_builtin(__builtin_amdgcn_tensor_load_to_lds)
#define HAVE_TDM 1
#else
#define HAVE_TDM 0
#endif
#if defined(__AMDGCN__) && __has_builtin(__builtin_amdgcn_global_load_async_to_lds_b128)
#define HAVE_ASYNC 1
#else
#define HAVE_ASYNC 0
#endif
#if defined(__AMDGCN__) && __has_builtin(__builtin_amdgcn_s_wait_asynccnt)
#define WAIT_ASYNC() __builtin_amdgcn_s_wait_asynccnt(0)
#elif defined(__AMDGCN__)
#define WAIT_ASYNC() asm volatile("s_wait_asynccnt 0x0" ::: "memory")
#else
#define WAIT_ASYNC() ((void)0)
#endif

#if HAVE_ASYNC
__device__ inline void async_ld_b128(const void* gsrc, const void* ldst) {
    gas_i32x4* gp = (gas_i32x4*)(unsigned long long)gsrc;
    las_i32x4* lp = (las_i32x4*)(unsigned)(unsigned long long)ldst;  // low 32 bits = LDS offset
    __builtin_amdgcn_global_load_async_to_lds_b128(gp, lp, 0, 0);
}
#endif

#if HAVE_TDM
// Build a 2D tensor D# (group0/group1) and issue TENSOR_LOAD_TO_LDS.
// rows x row_elems f16 tile, source row stride row_stride_elems, dense into LDS.
__device__ inline void tdm_load_2d_f16(unsigned lds_byte_off, const void* gsrc,
                                       unsigned rows, unsigned row_elems,
                                       unsigned row_stride_elems) {
    unsigned long long ga = (unsigned long long)gsrc;
    u32x4 g0;
    g0[0] = 1u;                                             // count=1, user descriptor
    g0[1] = lds_byte_off;                                   // lds_addr
    g0[2] = (unsigned)(ga & 0xFFFFFFFFull);                 // global_addr[31:0]
    g0[3] = (unsigned)((ga >> 32) & 0x01FFFFFFull)          // global_addr[56:32]
          | (2u << 30);                                     // type = 2 ("image")
    i32x8 g1;
    g1[0] = (int)(1u << 16);                                // data_size=1 -> 2 bytes
    g1[1] = (int)((row_elems & 0xFFFFu) << 16);             // tensor_dim0[15:0]
    g1[2] = (int)((row_elems >> 16) | ((rows & 0xFFFFu) << 16)); // dim0 hi | tensor_dim1 lo
    g1[3] = (int)((rows >> 16) | ((row_elems & 0xFFFFu) << 16)); // dim1 hi | tile_dim0
    g1[4] = (int)(rows & 0xFFFFu);                          // tile_dim1 (tile_dim2=0)
    g1[5] = (int)row_stride_elems;                          // tensor_dim0_stride[31:0]
    g1[6] = 0;
    g1[7] = 0;
    i32x4 z4 = {0, 0, 0, 0};
#if __clang_major__ >= 23
    i32x8 z8 = {0, 0, 0, 0, 0, 0, 0, 0};
    __builtin_amdgcn_tensor_load_to_lds(g0, g1, z4, z4, z8, 0);
#else
    __builtin_amdgcn_tensor_load_to_lds(g0, g1, z4, z4, 0);
#endif
}
#endif

// ---------- wave32 cross-lane helpers (ds_bpermute) ----------
__device__ inline float shfl_xor_f(float v, int off) {
    int i = __builtin_bit_cast(int, v);
    int src = (((int)(threadIdx.x & 31) ^ off) << 2);
    int r = __builtin_amdgcn_ds_bpermute(src, i);
    return __builtin_bit_cast(float, r);
}
__device__ inline int shfl_xor_i(int v, int off) {
    int src = (((int)(threadIdx.x & 31) ^ off) << 2);
    return __builtin_amdgcn_ds_bpermute(src, v);
}
__device__ inline float wred_max(float v){ for (int o=16;o;o>>=1) v = fmaxf(v, shfl_xor_f(v,o)); return v; }
__device__ inline float wred_sum(float v){ for (int o=16;o;o>>=1) v += shfl_xor_f(v,o); return v; }
__device__ inline int   wred_min(int   v){ for (int o=16;o;o>>=1) { int t = shfl_xor_i(v,o); v = t < v ? t : v; } return v; }

// ---------- f32 -> f16 convert ----------
__global__ void cvt_f32_f16(const float* __restrict__ s, _Float16* __restrict__ d, int n) {
    int i = blockIdx.x * blockDim.x + threadIdx.x;
    if (i < n) d[i] = (_Float16)s[i];
}

// ---------- C[M,N] = A[M,K] @ B[N,K]^T + bias[N] ----------
// Block = 8 waves sharing one N (weight) tile, 8 consecutive M tiles -> weight
// fragment loads are identical across waves (WGP$/L2 hits).
template<bool F32OUT>
__global__ void gemm_at_wmma(const _Float16* __restrict__ A, const _Float16* __restrict__ Bm,
                             const float* __restrict__ bias, void* __restrict__ Cv,
                             int M, int N, int K) {
    const int wave = threadIdx.x >> 5;
    const int lane = threadIdx.x & 31;
    const int hi   = lane >> 4;
    const int l15  = lane & 15;
    const int mgroups = M >> 7;                 // groups of 128 rows
    const int mg = blockIdx.x % mgroups;
    const int ng = blockIdx.x / mgroups;
    const int m0 = mg * 128 + wave * 16;
    const int n0 = ng * 16;

    const _Float16* Ap = A  + (size_t)(m0 + l15) * K + (hi ? 8 : 0);
    const _Float16* Bp = Bm + (size_t)(n0 + l15) * K + (hi ? 8 : 0);

    v8f c = {};
    for (int k0 = 0; k0 < K; k0 += 32) {
        __builtin_prefetch((const void*)(Ap + k0 + 256), 0, 0);   // global_prefetch_b8
        v8h a0 = *(const v8h*)(Ap + k0);
        v8h a1 = *(const v8h*)(Ap + k0 + 16);
        v8h b0 = *(const v8h*)(Bp + k0);
        v8h b1 = *(const v8h*)(Bp + k0 + 16);
        v16h a, b;
#pragma unroll
        for (int j = 0; j < 8; ++j) { a[j]=a0[j]; a[8+j]=a1[j]; b[j]=b0[j]; b[8+j]=b1[j]; }
        c = __builtin_amdgcn_wmma_f32_16x16x32_f16(false, a, false, b, (short)0, c, false, false);
    }
    const float bv = bias[n0 + l15];
#pragma unroll
    for (int r = 0; r < 8; ++r) {
        int row = m0 + r + hi * 8;
        int col = n0 + l15;
        float v = c[r] + bv;
        if (F32OUT) ((float*)   Cv)[(size_t)row * N + col] = v;
        else        ((_Float16*)Cv)[(size_t)row * N + col] = (_Float16)v;
    }
}

// ---------- fused attention: QK^T -> softmax -> entropy top-k -> renorm -> PV ----------
__global__ void attn_sparse_kernel(const _Float16* __restrict__ Qh, const _Float16* __restrict__ Kh,
                                   const _Float16* __restrict__ Vh, _Float16* __restrict__ AO) {
    extern __shared__ char smem[];
    float*    S    = (float*)(smem + OFF_S);        // [16][SPITCH] f32 scores
    _Float16* Ps   = (_Float16*)(smem + OFF_PS);    // [16][SPITCH] f16 sparse probs
    _Float16* Qs   = (_Float16*)(smem + OFF_QS);    // [16][64]
    float*    Oacc = (float*)(smem + OFF_OACC);     // [16][64]

    const int wave = threadIdx.x >> 5;
    const int lane = threadIdx.x & 31;
    const int hi   = lane >> 4;
    const int l15  = lane & 15;

    const int idx = blockIdx.x;
    const int b   = idx / (NH * (SQ / 16));
    const int rem = idx % (NH * (SQ / 16));
    const int h   = rem / (SQ / 16);
    const int qt  = rem % (SQ / 16);
    const int gq  = b * SQ + qt * 16;
    const int hc  = h * DH;
    const size_t kvbase = (size_t)b * SKV * DE;

    // ---- stage Q tile (16x64 f16) in LDS; async-LDS path when available ----
#if HAVE_ASYNC
    if (threadIdx.x < 128) {
        int m = threadIdx.x >> 3, piece = threadIdx.x & 7;
        const _Float16* gp = Qh + (size_t)(gq + m) * DE + hc + piece * 8;
        _Float16* lp = Qs + threadIdx.x * 8;
        async_ld_b128(gp, lp);
    }
    WAIT_ASYNC();
#else
    for (int i = threadIdx.x; i < 16 * DH; i += blockDim.x) {
        int m = i >> 6, d = i & 63;
        Qs[i] = Qh[(size_t)(gq + m) * DE + hc + d];
    }
#endif
    __syncthreads();

    // ---- S = (Q @ K^T) * scale ; wave w owns KV tiles nt = 8w..8w+7 ----
    const float scale = 0.125f;
    for (int j = 0; j < 8; ++j) {
        int nt = wave * 8 + j;
        const _Float16* Kp = Kh + kvbase + (size_t)(nt * 16 + l15) * DE + hc + (hi ? 8 : 0);
        v8f c = {};
        for (int k0 = 0; k0 < DH; k0 += 32) {
            const _Float16* Aq = Qs + l15 * DH + k0 + (hi ? 8 : 0);
            v16h a, bm;
#pragma unroll
            for (int u = 0; u < 8; ++u) { a[u] = Aq[u]; a[8+u] = Aq[16+u]; }
            v8h b0 = *(const v8h*)(Kp + k0);
            v8h b1 = *(const v8h*)(Kp + k0 + 16);
#pragma unroll
            for (int u = 0; u < 8; ++u) { bm[u] = b0[u]; bm[8+u] = b1[u]; }
            c = __builtin_amdgcn_wmma_f32_16x16x32_f16(false, a, false, bm, (short)0, c, false, false);
        }
#pragma unroll
        for (int r = 0; r < 8; ++r)
            S[(r + hi * 8) * SPITCH + nt * 16 + l15] = c[r] * scale;
    }
    __syncthreads();

    // ---- softmax + entropy + dynamic top-k + renorm; wave w owns rows 2w, 2w+1 ----
    for (int rr = 0; rr < 2; ++rr) {
        const int m = wave * 2 + rr;
        const float* Srow = S + m * SPITCH;
        float v[32];
        float mx = -3.4e38f;
#pragma unroll
        for (int i = 0; i < 32; ++i) { v[i] = Srow[i * 32 + lane]; mx = fmaxf(mx, v[i]); }
        mx = wred_max(mx);
        float sum = 0.f;
#pragma unroll
        for (int i = 0; i < 32; ++i) { v[i] = __expf(v[i] - mx); sum += v[i]; }
        sum = wred_sum(sum);
        const float inv = 1.f / sum;
        float ent = 0.f;
#pragma unroll
        for (int i = 0; i < 32; ++i) { v[i] *= inv; ent += -v[i] * __logf(v[i] + 1e-8f); }
        ent = wred_sum(ent);
        int topk = (int)(32.f * (1.f - ent));    // trunc toward zero, like .int()
        topk = topk < 1 ? 1 : (topk > 32 ? 32 : topk);

        // extract 32 maxima via register mask + cross-wave argmax; k-th is threshold
        unsigned removed = 0u;
        float thr = 0.f;
        for (int it = 0; it < 32; ++it) {
            float best = -1.f; int bj = 0;
#pragma unroll
            for (int i = 0; i < 32; ++i) {
                bool live = !((removed >> i) & 1u);
                if (live && v[i] > best) { best = v[i]; bj = i; }
            }
            float wmax = wred_max(best);
            int cand = (best == wmax) ? lane : 64;
            int winner = wred_min(cand);
            if (lane == winner) removed |= (1u << bj);
            if (it == topk - 1) thr = wmax;
        }
        float ksum = 0.f;
#pragma unroll
        for (int i = 0; i < 32; ++i) { float s = (v[i] >= thr) ? v[i] : 0.f; v[i] = s; ksum += s; }
        ksum = wred_sum(ksum);
        const float rn = 1.f / (ksum + 1e-8f);
#pragma unroll
        for (int i = 0; i < 32; ++i)
            Ps[m * SPITCH + i * 32 + lane] = (_Float16)(v[i] * rn);
    }
    for (int i = threadIdx.x; i < 16 * DH; i += blockDim.x) Oacc[i] = 0.f;
    __syncthreads();

    // ---- O = P @ V ; wave w owns K slice [128w, 128w+128); V staged via TDM ----
    _Float16* Vs = (_Float16*)(smem + OFF_VS) + wave * (32 * DH);
    const unsigned vs_lds_off = (unsigned)(unsigned long long)(void*)Vs;  // low 32 = LDS byte offset
    v8f c[4] = {};
    for (int ks = 0; ks < 4; ++ks) {
        const int k0 = wave * 128 + ks * 32;
        const _Float16* Vg = Vh + kvbase + (size_t)k0 * DE + hc;   // 32 rows x 64 f16, stride DE
#if HAVE_TDM
        tdm_load_2d_f16(vs_lds_off, (const void*)Vg, 32, DH, DE);
        __builtin_amdgcn_s_wait_tensorcnt(0);
#elif HAVE_ASYNC
#pragma unroll
        for (int j = 0; j < 8; ++j) {                // 32*64*2B = 256 x 16B chunks
            int chunk = j * 32 + lane;
            int row = chunk >> 3, piece = chunk & 7;
            async_ld_b128(Vg + (size_t)row * DE + piece * 8, Vs + chunk * 8);
        }
        WAIT_ASYNC();
#else
#pragma unroll
        for (int j = 0; j < 8; ++j) {
            int chunk = j * 32 + lane;
            int row = chunk >> 3, piece = chunk & 7;
            *(v8h*)(Vs + chunk * 8) = *(const v8h*)(Vg + (size_t)row * DE + piece * 8);
        }
#endif
        // consume staged tile for all 4 output column tiles
        const _Float16* Pp = Ps + l15 * SPITCH + k0 + (hi ? 16 : 0);
        v8h a0 = *(const v8h*)Pp;
        v8h a1 = *(const v8h*)(Pp + 8);
        v16h a;
#pragma unroll
        for (int u = 0; u < 8; ++u) { a[u] = a0[u]; a[8+u] = a1[u]; }
        const int kbB = hi ? 16 : 0;
#pragma unroll
        for (int nt = 0; nt < 4; ++nt) {
            v16h bm;
#pragma unroll
            for (int u = 0; u < 16; ++u) bm[u] = Vs[(kbB + u) * DH + nt * 16 + l15];
            c[nt] = __builtin_amdgcn_wmma_f32_16x16x32_f16(false, a, false, bm, (short)0, c[nt], false, false);
        }
    }
#pragma unroll
    for (int nt = 0; nt < 4; ++nt)
#pragma unroll
        for (int r = 0; r < 8; ++r)
            atomicAdd(&Oacc[(r + hi * 8) * DH + nt * 16 + l15], c[nt][r]);
    __syncthreads();
    for (int i = threadIdx.x; i < 16 * DH; i += blockDim.x) {
        int m = i >> 6, d = i & 63;
        AO[(size_t)(gq + m) * DE + hc + d] = (_Float16)Oacc[i];
    }
}

extern "C" void kernel_launch(void* const* d_in, const int* in_sizes, int n_in,
                              void* d_out, int out_size, void* d_ws, size_t ws_size,
                              hipStream_t stream) {
    const float* x  = (const float*)d_in[0];
    const float* y  = (const float*)d_in[1];
    const float* wq = (const float*)d_in[2];
    const float* bq = (const float*)d_in[3];
    const float* wk = (const float*)d_in[4];
    const float* bk = (const float*)d_in[5];
    const float* wv = (const float*)d_in[6];
    const float* bv = (const float*)d_in[7];
    const float* wo = (const float*)d_in[8];
    const float* bo = (const float*)d_in[9];
    float* out = (float*)d_out;

    const int MQ  = BB * SQ;    // 16384
    const int MKV = BB * SKV;   // 8192

    char* ws = (char*)d_ws;
    size_t off = 0;
    auto carve = [&](size_t bytes) {
        char* p = ws + off;
        off += (bytes + 255) & ~(size_t)255;
        return p;
    };
    _Float16* xh  = (_Float16*)carve((size_t)MQ  * DE * 2);
    _Float16* yh  = (_Float16*)carve((size_t)MKV * DC * 2);
    _Float16* wqh = (_Float16*)carve((size_t)DE * DE * 2);
    _Float16* wkh = (_Float16*)carve((size_t)DE * DC * 2);
    _Float16* wvh = (_Float16*)carve((size_t)DE * DC * 2);
    _Float16* woh = (_Float16*)carve((size_t)DE * DE * 2);
    _Float16* Qh  = (_Float16*)carve((size_t)MQ  * DE * 2);
    _Float16* Kh  = (_Float16*)carve((size_t)MKV * DE * 2);
    _Float16* Vh  = (_Float16*)carve((size_t)MKV * DE * 2);
    _Float16* AOh = (_Float16*)carve((size_t)MQ  * DE * 2);

    auto cvt = [&](const float* s, _Float16* d, int n) {
        cvt_f32_f16<<<(n + 255) / 256, 256, 0, stream>>>(s, d, n);
    };
    cvt(x,  xh,  MQ  * DE);
    cvt(y,  yh,  MKV * DC);
    cvt(wq, wqh, DE * DE);
    cvt(wk, wkh, DE * DC);
    cvt(wv, wvh, DE * DC);
    cvt(wo, woh, DE * DE);

    // projections: C = A @ W^T + b  (block = 8 M-tiles sharing one weight tile)
    {
        int blocks = (MQ / 128) * (DE / 16);
        gemm_at_wmma<false><<<blocks, 256, 0, stream>>>(xh, wqh, bq, Qh, MQ, DE, DE);
    }
    {
        int blocks = (MKV / 128) * (DE / 16);
        gemm_at_wmma<false><<<blocks, 256, 0, stream>>>(yh, wkh, bk, Kh, MKV, DE, DC);
        gemm_at_wmma<false><<<blocks, 256, 0, stream>>>(yh, wvh, bv, Vh, MKV, DE, DC);
    }

    // fused attention with dynamic sparse top-k
    {
        int blocks = BB * NH * (SQ / 16);               // 8192
        attn_sparse_kernel<<<blocks, 256, SMEM_TOTAL, stream>>>(Qh, Kh, Vh, AOh);
    }

    // output projection (f32 out to d_out)
    {
        int blocks = (MQ / 128) * (DE / 16);
        gemm_at_wmma<true><<<blocks, 256, 0, stream>>>(AOh, woh, bo, (void*)out, MQ, DE, DE);
    }
}
